// GraspSampler_67190468378987
// MI455X (gfx1250) — compile-verified, matrix operand order checked by reading, not run on hardware
//
#include <hip/hip_runtime.h>

// GraspSampler (PointNet++ encoder) for MI455X / gfx1250, wave32.
// All matmuls (shared-MLP 1x1 convs + FC layers) run through
// V_WMMA_F32_16X16X32_F16 (f16 in, f32 accumulate). Activations are stored
// f16 to halve HBM traffic (23.3 TB/s); BN batch statistics are recomputed
// deterministically per layer (no float atomics -> replay-stable).

typedef _Float16 half_t;
typedef __attribute__((ext_vector_type(16))) _Float16 v16h;
typedef __attribute__((ext_vector_type(8)))  float    v8f;

#define BN_EPS 1e-5f

// K-offset within a 16x32 f16 WMMA fragment (ISA 7.12.2, 16-bit A 16x32):
// VGPR v (=e>>1), pair bit (=e&1); half-wave hf selects K+0..7 vs K+8..15,
// VGPRs 4..7 cover K=16..31 with the same half-wave split.
__device__ __forceinline__ int wmma_kofs(int hf, int e) {
  int v = e >> 1, pr = e & 1;
  return (v < 4) ? (hf * 8 + v * 2 + pr) : (16 + hf * 8 + (v - 4) * 2 + pr);
}

// ---------------------------------------------------------------- features
// feats (B,6,N): rows 0-2 xyz^T, 3-4 broadcast z, 5 query flag (n==N-1)
__global__ void k_feats(const float* __restrict__ xyz, const float* __restrict__ z,
                        float* __restrict__ feats, int B, int N) {
  int gid = blockIdx.x * blockDim.x + threadIdx.x;
  if (gid >= B * N) return;
  int b = gid / N, n = gid % N;
  const float* p = xyz + ((size_t)b * N + n) * 3;
  float* f = feats + (size_t)b * 6 * N;
  f[(size_t)0 * N + n] = p[0];
  f[(size_t)1 * N + n] = p[1];
  f[(size_t)2 * N + n] = p[2];
  f[(size_t)3 * N + n] = z[b * 2 + 0];
  f[(size_t)4 * N + n] = z[b * 2 + 1];
  f[(size_t)5 * N + n] = (n == N - 1) ? 1.0f : 0.0f;
}

// ------------------------------------------------------------- weight prep
// f32 (O x Cin) row-major -> f16 (O x Cpad), zero-padded K for WMMA steps.
__global__ void k_convert_w(const float* __restrict__ w, half_t* __restrict__ out,
                            int O, int Cin, int Cpad) {
  int i = blockIdx.x * blockDim.x + threadIdx.x;
  if (i >= O * Cpad) return;
  int o = i / Cpad, c = i % Cpad;
  out[i] = (half_t)((c < Cin) ? w[(size_t)o * Cin + c] : 0.0f);
}

// ----------------------------------------------------------------- FPS
// One workgroup per batch; dists[] lives in LDS. Matches CUDA-style FPS:
// start at index 0, argmax picks the first (lowest) index on ties.
__global__ void k_fps(const float* __restrict__ xyz, int N, int npoint,
                      int* __restrict__ outIdx) {
  extern __shared__ float smem[];
  float* dists = smem;           // N floats
  float* rv = smem + N;          // 256 floats
  int* ri = (int*)(rv + 256);    // 256 ints
  __shared__ float s_pt[3];
  __shared__ int s_last;
  int b = blockIdx.x, tid = threadIdx.x;
  const float* pts = xyz + (size_t)b * N * 3;
  for (int n = tid; n < N; n += 256) dists[n] = 1e10f;
  if (tid == 0) { s_last = 0; outIdx[(size_t)b * npoint] = 0; }
  __syncthreads();
  for (int i = 1; i < npoint; ++i) {
    if (tid < 3) s_pt[tid] = pts[(size_t)s_last * 3 + tid];
    __syncthreads();
    float bx = s_pt[0], by = s_pt[1], bz = s_pt[2];
    float bestv = -1.0f; int besti = N - 1;
    for (int n = tid; n < N; n += 256) {
      float dx = pts[n * 3 + 0] - bx;
      float dy = pts[n * 3 + 1] - by;
      float dz = pts[n * 3 + 2] - bz;
      float d = dx * dx + dy * dy + dz * dz;
      float dm = fminf(dists[n], d);
      dists[n] = dm;
      if (dm > bestv) { bestv = dm; besti = n; }
    }
    rv[tid] = bestv; ri[tid] = besti;
    __syncthreads();
    for (int st = 128; st > 0; st >>= 1) {
      if (tid < st) {
        float ov = rv[tid + st]; int oi = ri[tid + st];
        if (ov > rv[tid] || (ov == rv[tid] && oi < ri[tid])) { rv[tid] = ov; ri[tid] = oi; }
      }
      __syncthreads();
    }
    if (tid == 0) { s_last = ri[0]; outIdx[(size_t)b * npoint + i] = ri[0]; }
    __syncthreads();
  }
}

__global__ void k_gather_centers(const float* __restrict__ xyz, const int* __restrict__ cidx,
                                 float* __restrict__ centers, int B, int N, int S) {
  int gid = blockIdx.x * blockDim.x + threadIdx.x;
  if (gid >= B * S) return;
  int b = gid / S, s = gid % S;
  int j = cidx[(size_t)b * S + s];
  const float* p = xyz + ((size_t)b * N + j) * 3;
  float* o = centers + ((size_t)b * S + s) * 3;
  o[0] = p[0]; o[1] = p[1]; o[2] = p[2];
}

// ---------------------------------------------------------------- ball query
// First K indices (ascending) with d2 < r2, padded with the first hit
// (centroid is a member point, so >=1 hit always exists).
__global__ void k_ball(const float* __restrict__ xyz, const float* __restrict__ centers,
                       int* __restrict__ idx, int B, int N, int S, int K, float r2) {
  int gid = blockIdx.x * blockDim.x + threadIdx.x;
  if (gid >= B * S) return;
  int b = gid / S, s = gid % S;
  const float* c = centers + ((size_t)b * S + s) * 3;
  const float* p = xyz + (size_t)b * N * 3;
  int* o = idx + ((size_t)b * S + s) * K;
  int cnt = 0;
  for (int j = 0; j < N && cnt < K; ++j) {
    float dx = p[j * 3 + 0] - c[0];
    float dy = p[j * 3 + 1] - c[1];
    float dz = p[j * 3 + 2] - c[2];
    if (dx * dx + dy * dy + dz * dz < r2) o[cnt++] = j;
  }
  int f = (cnt > 0) ? o[0] : 0;
  for (; cnt < K; ++cnt) o[cnt] = f;
}

// ----------------------------------------------------------------- grouping
// Build GEMM input X (Cpad x Ntot) f16, column n = ((b*S+s)*K + k).
// Rows 0..2: gathered xyz minus centroid; rows 3..3+C-1: gathered feats.
__global__ void k_group(const float* __restrict__ xyz, const float* __restrict__ feats,
                        const float* __restrict__ centers, const int* __restrict__ idx,
                        half_t* __restrict__ X, int B, int N, int S, int K, int C, int Cpad) {
  long long Ntot = (long long)B * S * K;
  long long n = (long long)blockIdx.x * blockDim.x + threadIdx.x;
  if (n >= Ntot) return;
  int k = (int)(n % K);
  long long t = n / K;
  int s = (int)(t % S);
  int b = (int)(t / S);
  int j = idx[((size_t)b * S + s) * K + k];
  const float* c = centers + ((size_t)b * S + s) * 3;
  const float* p = xyz + ((size_t)b * N + j) * 3;
  X[0 * Ntot + n] = (half_t)(p[0] - c[0]);
  X[1 * Ntot + n] = (half_t)(p[1] - c[1]);
  X[2 * Ntot + n] = (half_t)(p[2] - c[2]);
  const float* f = feats + (size_t)b * C * N;
  for (int cc = 0; cc < C; ++cc)
    X[(long long)(3 + cc) * Ntot + n] = (half_t)f[(size_t)cc * N + j];
  for (int r = 3 + C; r < Cpad; ++r)
    X[(long long)r * Ntot + n] = (half_t)0.0f;
}

// group-all for SA3: column n = b*S+s; rows 0-2 xyz (NOT centered), 3.. feats.
__global__ void k_group_all(const float* __restrict__ xyz, const float* __restrict__ feats,
                            half_t* __restrict__ X, int B, int S, int C, int Cpad) {
  long long Ntot = (long long)B * S;
  long long n = (long long)blockIdx.x * blockDim.x + threadIdx.x;
  if (n >= Ntot) return;
  int s = (int)(n % S);
  int b = (int)(n / S);
  const float* p = xyz + ((size_t)b * S + s) * 3;
  X[0 * Ntot + n] = (half_t)p[0];
  X[1 * Ntot + n] = (half_t)p[1];
  X[2 * Ntot + n] = (half_t)p[2];
  const float* f = feats + (size_t)b * C * S;
  for (int cc = 0; cc < C; ++cc)
    X[(long long)(3 + cc) * Ntot + n] = (half_t)f[(size_t)cc * S + s];
  for (int r = 3 + C; r < Cpad; ++r)
    X[(long long)r * Ntot + n] = (half_t)0.0f;
}

// ------------------------------------------------------------------- GEMM
// Y(MxN) = A(MxK) * X(KxN), all f16 in memory, f32 WMMA accumulate.
// One wave per 16x16 output tile; K stepped by 32 via v_wmma_f32_16x16x32_f16.
__global__ void k_gemm_f16(const half_t* __restrict__ A, const half_t* __restrict__ X,
                           half_t* __restrict__ Y, int M, long long N, int K) {
  long long wid = (long long)blockIdx.x * (blockDim.x >> 5) + (threadIdx.x >> 5);
  long long tilesN = N >> 4;
  long long mt = wid / tilesN;
  long long nt = wid % tilesN;
  if (mt >= (long long)(M >> 4)) return;   // wave-uniform exit keeps EXEC all-1s
  int lane = threadIdx.x & 31;
  int l16 = lane & 15, hf = lane >> 4;
  v8f c = {};
  long long rowA = mt * 16 + l16;
  long long colB = nt * 16 + l16;
  for (int k0 = 0; k0 < K; k0 += 32) {
    v16h a, b;
#pragma unroll
    for (int e = 0; e < 16; ++e) {
      int ko = wmma_kofs(hf, e);
      a[e] = A[rowA * K + (k0 + ko)];
      b[e] = X[(long long)(k0 + ko) * N + colB];
    }
    c = __builtin_amdgcn_wmma_f32_16x16x32_f16(
        /*neg_a=*/false, a, /*neg_b=*/false, b,
        /*c_mod=*/(short)0, c, /*reuse_a=*/false, /*reuse_b=*/false);
  }
#pragma unroll
  for (int r = 0; r < 8; ++r) {
    long long row = mt * 16 + r + 8 * hf;   // C/D layout: lanes 16-31 hold M=r+8
    Y[row * N + colB] = (half_t)c[r];
  }
}

// per-channel sum / sumsq for BatchNorm (deterministic tree reduction)
__global__ void k_rowstats(const half_t* __restrict__ Y, float* __restrict__ sum,
                           float* __restrict__ sumsq, long long N) {
  int row = blockIdx.x;
  const half_t* p = Y + (long long)row * N;
  float s = 0.0f, s2 = 0.0f;
  for (long long i = threadIdx.x; i < N; i += blockDim.x) {
    float v = (float)p[i];
    s += v; s2 += v * v;
  }
  __shared__ float sh[256], sh2[256];
  sh[threadIdx.x] = s; sh2[threadIdx.x] = s2;
  __syncthreads();
  for (int st = 128; st > 0; st >>= 1) {
    if ((int)threadIdx.x < st) {
      sh[threadIdx.x] += sh[threadIdx.x + st];
      sh2[threadIdx.x] += sh2[threadIdx.x + st];
    }
    __syncthreads();
  }
  if (threadIdx.x == 0) { sum[row] = sh[0]; sumsq[row] = sh2[0]; }
}

// BN (batch stats) + ReLU, writes next layer's f16 input (rows padded to Mpad)
__global__ void k_bnrelu(const half_t* __restrict__ Y, const float* __restrict__ sum,
                         const float* __restrict__ sumsq, const float* __restrict__ g,
                         const float* __restrict__ b, half_t* __restrict__ Xout,
                         int M, int Mpad, long long N, float inv_n) {
  long long tot = (long long)Mpad * N;
  long long i = (long long)blockIdx.x * blockDim.x + threadIdx.x;
  if (i >= tot) return;
  int r = (int)(i / N);
  long long col = i % N;
  float v = 0.0f;
  if (r < M) {
    float mean = sum[r] * inv_n;
    float var = fmaxf(sumsq[r] * inv_n - mean * mean, 0.0f);
    float y = (float)Y[(long long)r * N + col];
    v = (y - mean) * rsqrtf(var + BN_EPS) * g[r] + b[r];
    v = fmaxf(v, 0.0f);
  }
  Xout[i] = (half_t)v;
}

// BN + ReLU + max over the K group dimension -> out (B, M, S) f32
__global__ void k_bnrelu_max(const half_t* __restrict__ Y, const float* __restrict__ sum,
                             const float* __restrict__ sumsq, const float* __restrict__ g,
                             const float* __restrict__ b, float* __restrict__ out,
                             int M, int B, int S, int K, float inv_n) {
  long long N = (long long)B * S * K;
  int gid = blockIdx.x * blockDim.x + threadIdx.x;
  int tot = B * M * S;
  if (gid >= tot) return;
  int s = gid % S;
  int t = gid / S;
  int o = t % M;
  int bb = t / M;
  float mean = sum[o] * inv_n;
  float inv = rsqrtf(fmaxf(sumsq[o] * inv_n - mean * mean, 0.0f) + BN_EPS) * g[o];
  float bo = b[o];
  const half_t* p = Y + (long long)o * N + ((long long)bb * S + s) * K;
  float m = -1e30f;
  for (int k = 0; k < K; ++k) {
    float v = ((float)p[k] - mean) * inv + bo;
    v = fmaxf(v, 0.0f);
    m = fmaxf(m, v);
  }
  out[((size_t)bb * M + o) * S + s] = m;
}

// f32 (Ncols x K) row-major -> f16 (Kpad x Ncols) for the FC GEMMs
__global__ void k_t_convert(const float* __restrict__ in, half_t* __restrict__ out,
                            int Ncols, int K, int Kpad) {
  int i = blockIdx.x * blockDim.x + threadIdx.x;
  if (i >= Kpad * Ncols) return;
  int r = i / Ncols, c = i % Ncols;
  out[i] = (half_t)((r < K) ? in[(size_t)c * K + r] : 0.0f);
}

// q/t heads: x is f16 (C x B). One thread per output scalar.
__global__ void k_heads(const half_t* __restrict__ x, const float* __restrict__ wq,
                        const float* __restrict__ bq, const float* __restrict__ wt,
                        const float* __restrict__ bt, float* __restrict__ out,
                        int B, int C) {
  int gid = blockIdx.x * blockDim.x + threadIdx.x;
  if (gid >= B * 7) return;
  int b = gid / 7, j = gid % 7;
  const float* w = (j < 4) ? (wq + (size_t)j * C) : (wt + (size_t)(j - 4) * C);
  float acc = (j < 4) ? bq[j] : bt[j - 4];
  for (int c = 0; c < C; ++c) acc += (float)x[(size_t)c * B + b] * w[c];
  out[gid] = acc;
}

__global__ void k_qnorm(float* __restrict__ out, int B) {
  int b = blockIdx.x * blockDim.x + threadIdx.x;
  if (b >= B) return;
  float* q = out + (size_t)b * 7;
  float n = sqrtf(q[0] * q[0] + q[1] * q[1] + q[2] * q[2] + q[3] * q[3]);
  n = fmaxf(n, 1e-12f);
  q[0] /= n; q[1] /= n; q[2] /= n; q[3] /= n;
}

// =========================================================================
extern "C" void kernel_launch(void* const* d_in, const int* in_sizes, int n_in,
                              void* d_out, int out_size, void* d_ws, size_t ws_size,
                              hipStream_t stream) {
  (void)in_sizes; (void)n_in; (void)out_size; (void)ws_size;
  const int B = 32, N = 4096;

  // -------- inputs (setup_inputs() dict insertion order, recursive) --------
  const float* xyz = (const float*)d_in[0];
  const float* z   = (const float*)d_in[1];
  const float *sa_w[9], *sa_g[9], *sa_b[9];
  int p = 2;
  for (int i = 0; i < 9; ++i) {           // sa1 L0..2, sa2 L0..2, sa3 L0..2
    sa_w[i] = (const float*)d_in[p++];    // {'w','g','b'} insertion order
    sa_g[i] = (const float*)d_in[p++];
    sa_b[i] = (const float*)d_in[p++];
  }
  const float* fc1_w  = (const float*)d_in[p++];
  const float* fc1_b  = (const float*)d_in[p++]; (void)fc1_b;   // cancelled by BN
  const float* fc1_g  = (const float*)d_in[p++];
  const float* fc1_be = (const float*)d_in[p++];
  const float* fc2_w  = (const float*)d_in[p++];
  const float* fc2_b  = (const float*)d_in[p++]; (void)fc2_b;   // cancelled by BN
  const float* fc2_g  = (const float*)d_in[p++];
  const float* fc2_be = (const float*)d_in[p++];
  const float* q_w = (const float*)d_in[p++];
  const float* q_b = (const float*)d_in[p++];
  const float* t_w = (const float*)d_in[p++];
  const float* t_b = (const float*)d_in[p++];

  // ------------------------------ workspace --------------------------------
  char* ws = (char*)d_ws;
  auto alloc = [&](size_t bytes) -> char* {
    char* r = ws;
    ws += (bytes + 255) & ~(size_t)255;
    return r;
  };
  float* feats0 = (float*)alloc((size_t)B * 6 * N * 4);
  int*   cidx1  = (int*)alloc((size_t)B * 512 * 4);
  float* nxyz1  = (float*)alloc((size_t)B * 512 * 3 * 4);
  int*   idx1   = (int*)alloc((size_t)B * 512 * 64 * 4);
  int*   cidx2  = (int*)alloc((size_t)B * 32 * 4);
  float* nxyz2  = (float*)alloc((size_t)B * 32 * 3 * 4);
  int*   idx2   = (int*)alloc((size_t)B * 32 * 128 * 4);
  float* f1     = (float*)alloc((size_t)B * 128 * 512 * 4);
  float* f2     = (float*)alloc((size_t)B * 256 * 32 * 4);
  float* f3     = (float*)alloc((size_t)B * 512 * 4);
  float* stats  = (float*)alloc((size_t)2048 * 4);
  float* ssum = stats;
  float* ssq  = stats + 1024;
  // converted / padded f16 weights (9 conv layers + fc1 + fc2)
  const int WO[11]   = {64, 64, 128, 128, 128, 256, 256, 256, 512, 1024, 1024};
  const int WCin[11] = {9, 64, 64, 131, 128, 128, 259, 256, 256, 512, 1024};
  const int WCp[11]  = {32, 64, 64, 160, 128, 128, 288, 256, 256, 512, 1024};
  half_t* Wc[11];
  for (int i = 0; i < 11; ++i) Wc[i] = (half_t*)alloc((size_t)WO[i] * WCp[i] * 2);
  const long long NT1 = (long long)B * 512 * 64;   // 1,048,576 columns
  const long long NT2 = (long long)B * 32 * 128;   //   131,072 columns
  const long long NT3 = (long long)B * 32;         //     1,024 columns
  half_t* actA = (half_t*)alloc((size_t)160 * NT1 * 2);   // ping (inputs)
  half_t* actB = (half_t*)alloc((size_t)128 * NT1 * 2);   // pong (raw GEMM out)

  auto gemm = [&](const half_t* A, const half_t* X, half_t* Y, int M, long long Nn, int K) {
    long long waves = (long long)(M / 16) * (Nn / 16);
    int blocks = (int)((waves + 7) / 8);
    k_gemm_f16<<<blocks, 256, 0, stream>>>(A, X, Y, M, Nn, K);
    k_rowstats<<<M, 256, 0, stream>>>(Y, ssum, ssq, Nn);
  };
  auto bnrelu = [&](const half_t* Y, const float* g, const float* b, half_t* X,
                    int M, int Mpad, long long Nn) {
    long long tot = (long long)Mpad * Nn;
    k_bnrelu<<<(int)((tot + 255) / 256), 256, 0, stream>>>(Y, ssum, ssq, g, b, X, M, Mpad, Nn,
                                                           1.0f / (float)Nn);
  };

  // ------------------------------ stage 0 ----------------------------------
  k_feats<<<(B * N + 255) / 256, 256, 0, stream>>>(xyz, z, feats0, B, N);
  for (int i = 0; i < 9; ++i)
    k_convert_w<<<(WO[i] * WCp[i] + 255) / 256, 256, 0, stream>>>(sa_w[i], Wc[i], WO[i], WCin[i], WCp[i]);
  k_convert_w<<<(WO[9] * WCp[9] + 255) / 256, 256, 0, stream>>>(fc1_w, Wc[9], 1024, 512, 512);
  k_convert_w<<<(WO[10] * WCp[10] + 255) / 256, 256, 0, stream>>>(fc2_w, Wc[10], 1024, 1024, 1024);

  // ------------------------------ SA1 --------------------------------------
  k_fps<<<B, 256, (size_t)N * 4 + 256 * 8, stream>>>(xyz, N, 512, cidx1);
  k_gather_centers<<<(B * 512 + 255) / 256, 256, 0, stream>>>(xyz, cidx1, nxyz1, B, N, 512);
  k_ball<<<(B * 512 + 255) / 256, 256, 0, stream>>>(xyz, nxyz1, idx1, B, N, 512, 64, 0.1f * 0.1f);
  k_group<<<(int)((NT1 + 255) / 256), 256, 0, stream>>>(xyz, feats0, nxyz1, idx1, actA,
                                                        B, N, 512, 64, 6, 32);
  gemm(Wc[0], actA, actB, 64, NT1, 32);
  bnrelu(actB, sa_g[0], sa_b[0], actA, 64, 64, NT1);
  gemm(Wc[1], actA, actB, 64, NT1, 64);
  bnrelu(actB, sa_g[1], sa_b[1], actA, 64, 64, NT1);
  gemm(Wc[2], actA, actB, 128, NT1, 64);
  k_bnrelu_max<<<(B * 128 * 512 + 255) / 256, 256, 0, stream>>>(actB, ssum, ssq, sa_g[2], sa_b[2],
                                                                f1, 128, B, 512, 64,
                                                                1.0f / (float)NT1);

  // ------------------------------ SA2 --------------------------------------
  k_fps<<<B, 256, (size_t)512 * 4 + 256 * 8, stream>>>(nxyz1, 512, 32, cidx2);
  k_gather_centers<<<(B * 32 + 255) / 256, 256, 0, stream>>>(nxyz1, cidx2, nxyz2, B, 512, 32);
  k_ball<<<(B * 32 + 255) / 256, 256, 0, stream>>>(nxyz1, nxyz2, idx2, B, 512, 32, 128,
                                                   0.04f * 0.04f);
  k_group<<<(int)((NT2 + 255) / 256), 256, 0, stream>>>(nxyz1, f1, nxyz2, idx2, actA,
                                                        B, 512, 32, 128, 128, 160);
  gemm(Wc[3], actA, actB, 128, NT2, 160);
  bnrelu(actB, sa_g[3], sa_b[3], actA, 128, 128, NT2);
  gemm(Wc[4], actA, actB, 128, NT2, 128);
  bnrelu(actB, sa_g[4], sa_b[4], actA, 128, 128, NT2);
  gemm(Wc[5], actA, actB, 256, NT2, 128);
  k_bnrelu_max<<<(B * 256 * 32 + 255) / 256, 256, 0, stream>>>(actB, ssum, ssq, sa_g[5], sa_b[5],
                                                               f2, 256, B, 32, 128,
                                                               1.0f / (float)NT2);

  // ------------------------------ SA3 (group all) --------------------------
  k_group_all<<<(int)((NT3 + 255) / 256), 256, 0, stream>>>(nxyz2, f2, actA, B, 32, 256, 288);
  gemm(Wc[6], actA, actB, 256, NT3, 288);
  bnrelu(actB, sa_g[6], sa_b[6], actA, 256, 256, NT3);
  gemm(Wc[7], actA, actB, 256, NT3, 256);
  bnrelu(actB, sa_g[7], sa_b[7], actA, 256, 256, NT3);
  gemm(Wc[8], actA, actB, 512, NT3, 256);
  k_bnrelu_max<<<(B * 512 + 255) / 256, 256, 0, stream>>>(actB, ssum, ssq, sa_g[8], sa_b[8],
                                                          f3, 512, B, 1, 32, 1.0f / (float)NT3);

  // ------------------------------ FC head ----------------------------------
  k_t_convert<<<(512 * 32 + 255) / 256, 256, 0, stream>>>(f3, actA, 32, 512, 512);
  gemm(Wc[9], actA, actB, 1024, 32, 512);
  bnrelu(actB, fc1_g, fc1_be, actA, 1024, 1024, 32);
  gemm(Wc[10], actA, actB, 1024, 32, 1024);
  bnrelu(actB, fc2_g, fc2_be, actA, 1024, 1024, 32);
  k_heads<<<(B * 7 + 255) / 256, 256, 0, stream>>>(actA, q_w, q_b, t_w, t_b,
                                                   (float*)d_out, B, 1024);
  k_qnorm<<<1, 32, 0, stream>>>((float*)d_out, B);
}
